// GraphConstructor_35673998360736
// MI455X (gfx1250) — matrix operand order repaired and checked by reading, not compiled, not run on hardware
//
#include <hip/hip_runtime.h>
#include <hip/hip_bf16.h>

typedef __attribute__((ext_vector_type(16))) _Float16 v16h;
typedef __attribute__((ext_vector_type(8)))  float    v8f;

#define Bn     16
#define Nn     320
#define TOTALn 4096
#define En     65536
#define DINn   8
#define DEn    128

// ---- output layout (flat f32 in d_out) ----
#define NODE_OFF  ((size_t)0)
#define NODE_SZ   ((size_t)Bn * Nn * DEn)                 // 655,360
#define EDGE_OFF  (NODE_OFF + NODE_SZ)
#define EDGE_SZ   ((size_t)Bn * Nn * Nn * DEn)            // 209,715,200
#define MASK_OFF  (EDGE_OFF + EDGE_SZ)
#define MASK_SZ   ((size_t)Bn * Nn * Nn)                  // 1,638,400
#define NM_OFF    (MASK_OFF + MASK_SZ)

// ---- workspace layout (bytes) ----
#define DENSE_E_OFF ((size_t)0)                             // 16*320*320*8 f32 = 52,428,800 B
#define DENSE_X_OFF ((size_t)52428800)                      // 16*320*8 f32 (contiguous after dense_e)
#define STARTS_OFF  ((size_t)52592640)
#define COUNTS_OFF  ((size_t)52592704)
#define BPACK_OFF   ((size_t)52592768)                      // 8*32*16 halves = 8192 B
#define BCOMB_OFF   ((size_t)52600960)                      // 128 f32 (kept for reference/debug)
#define WNC_OFF     ((size_t)52601472)                      // 128*8 f32
#define BNC_OFF     ((size_t)52605568)                      // 128 f32

__global__ void zero_f4_kernel(float4* __restrict__ p, int n4) {
    for (int i = blockIdx.x * blockDim.x + threadIdx.x; i < n4; i += gridDim.x * blockDim.x)
        p[i] = float4{0.f, 0.f, 0.f, 0.f};
}

// counts per graph + exclusive-scan starts (single block)
__global__ void starts_kernel(const int* __restrict__ node_batch,
                              int* __restrict__ starts, int* __restrict__ counts) {
    __shared__ int scnt[Bn];
    int tid = threadIdx.x;
    if (tid < Bn) scnt[tid] = 0;
    __syncthreads();
    for (int i = tid; i < TOTALn; i += blockDim.x)
        atomicAdd(&scnt[node_batch[i]], 1);
    __syncthreads();
    if (tid == 0) {
        int acc = 0;
        for (int b = 0; b < Bn; ++b) { starts[b] = acc; counts[b] = scnt[b]; acc += scnt[b]; }
    }
}

__global__ void scatter_x_kernel(const float* __restrict__ x, const int* __restrict__ node_batch,
                                 const int* __restrict__ starts, float* __restrict__ dense_x) {
    int i = blockIdx.x * blockDim.x + threadIdx.x;
    if (i >= TOTALn) return;
    int b = node_batch[i];
    int p = i - starts[b];
    float* dst = dense_x + ((size_t)b * Nn + p) * DINn;
    const float* src = x + (size_t)i * DINn;
#pragma unroll
    for (int k = 0; k < DINn; ++k) dst[k] = src[k];
}

__global__ void scatter_edges_kernel(const float* __restrict__ edge_attr,
                                     const int* __restrict__ edge_index,
                                     const int* __restrict__ node_batch,
                                     const int* __restrict__ starts,
                                     float* __restrict__ dense_e) {
    int e = blockIdx.x * blockDim.x + threadIdx.x;
    if (e >= En) return;
    int row = edge_index[e];
    int col = edge_index[En + e];
    int b = node_batch[row];
    int s = starts[b];
    int r = row - s;
    int c = col - s;
    float* dst = dense_e + (((size_t)b * Nn + r) * Nn + c) * DINn;
    const float* src = edge_attr + (size_t)e * DINn;
#pragma unroll
    for (int k = 0; k < DINn; ++k) atomicAdd(&dst[k], src[k]);
}

// Fold the two linear layers (edge and node paths). Pack [W_comb^T ; b_comb] as a
// 32x128 f16 B matrix (K=0..23 weights, K=24 bias row, K=25..31 zero) in the wave32
// WMMA 16-bit B-matrix per-lane layout:
//   lane L: N = L%16, g = L/16; half slot s (0..15) <-> K = (s<8) ? g*8 + s : 16 + g*8 + (s-8)
// The GEMM A-matrix carries 1.0 in its K=24 slot, so D = A@B includes the bias.
__global__ void prep_weights_kernel(const float* __restrict__ w_weight_w, const float* __restrict__ w_weight_b,
                                    const float* __restrict__ w_edge_w,   const float* __restrict__ w_edge_b,
                                    const float* __restrict__ w_bias_w,   const float* __restrict__ w_bias_b,
                                    const float* __restrict__ w_node_w,   const float* __restrict__ w_node_b,
                                    _Float16* __restrict__ bpack, float* __restrict__ b_comb,
                                    float* __restrict__ wn_comb, float* __restrict__ bn_comb) {
    __shared__ float Wc[DEn][24];
    __shared__ float sbias[DEn];
    int l = threadIdx.x;               // 0..127 = output feature
    float acc[24];
#pragma unroll
    for (int d = 0; d < 24; ++d) acc[d] = 0.f;
    float accb = 0.f;
    for (int k = 0; k < DEn; ++k) {
        float we = w_edge_w[(size_t)l * DEn + k];
        accb += we * w_weight_b[k];
#pragma unroll
        for (int d = 0; d < 24; ++d) acc[d] += we * w_weight_w[(size_t)k * 24 + d];
    }
    float bc = accb + w_edge_b[l];
    b_comb[l] = bc;
    sbias[l]  = bc;
#pragma unroll
    for (int d = 0; d < 24; ++d) Wc[l][d] = acc[d];

    float accn[DINn];
#pragma unroll
    for (int d = 0; d < DINn; ++d) accn[d] = 0.f;
    float accnb = 0.f;
    for (int k = 0; k < DEn; ++k) {
        float wn = w_node_w[(size_t)l * DEn + k];
        accnb += wn * w_bias_b[k];
#pragma unroll
        for (int d = 0; d < DINn; ++d) accn[d] += wn * w_bias_w[(size_t)k * DINn + d];
    }
    bn_comb[l] = accnb + w_node_b[l];
#pragma unroll
    for (int d = 0; d < DINn; ++d) wn_comb[(size_t)l * DINn + d] = accn[d];

    __syncthreads();
    // bpack[(nt*32 + lane)*16 + h] ; B[k][n] = W_comb[n][k], B[24][n] = bias[n]
    for (int idx = l; idx < 8 * 32 * 16; idx += blockDim.x) {
        int nt   = idx >> 9;
        int lane = (idx >> 4) & 31;
        int h    = idx & 15;
        int n    = nt * 16 + (lane & 15);
        int g    = lane >> 4;
        int k    = (h < 8) ? (g * 8 + h) : (16 + g * 8 + (h - 8));
        float v  = (k < 24) ? Wc[n][k] : ((k == 24) ? sbias[n] : 0.f);
        bpack[idx] = (_Float16)v;
    }
}

__global__ void mask_kernel(const float* __restrict__ dense_e, float* __restrict__ mask_out) {
    int idx = blockIdx.x * blockDim.x + threadIdx.x;   // B*N*N
    if (idx >= (int)MASK_SZ) return;
    int j = idx % Nn;
    int t = idx / Nn;
    int i = t % Nn;
    int b = t / Nn;
    const float* eij = dense_e + (((size_t)b * Nn + i) * Nn + j) * DINn;
    const float* eji = dense_e + (((size_t)b * Nn + j) * Nn + i) * DINn;
    float s1 = 0.f, s2 = 0.f;
#pragma unroll
    for (int k = 0; k < DINn; ++k) { s1 += eij[k]; s2 += eji[k]; }
    float mv = ((s1 != 0.f) || (s2 != 0.f)) ? 1.f : 0.f;
    __builtin_nontemporal_store(mv, &mask_out[idx]);
}

__global__ void node_mask_kernel(const int* __restrict__ counts, float* __restrict__ nm_out) {
    int idx = blockIdx.x * blockDim.x + threadIdx.x;   // B*N
    if (idx >= Bn * Nn) return;
    int p = idx % Nn;
    int b = idx / Nn;
    nm_out[idx] = (p < counts[b]) ? 1.f : 0.f;
}

__global__ void node_kernel(const float* __restrict__ dense_x, const float* __restrict__ wn_comb,
                            const float* __restrict__ bn_comb, float* __restrict__ node_out) {
    int idx = blockIdx.x * blockDim.x + threadIdx.x;   // B*N*128
    if (idx >= (int)NODE_SZ) return;
    int l   = idx & 127;
    int row = idx >> 7;
    const float* xr = dense_x + (size_t)row * DINn;
    const float* w  = wn_comb + (size_t)l * DINn;
    float acc = bn_comb[l];
#pragma unroll
    for (int k = 0; k < DINn; ++k) acc += w[k] * xr[k];
    node_out[idx] = acc;
}

// Fused edge GEMM: edge[r, 0:128] = [e_cat_row(r) | 1] @ [Wcomb^T ; b_comb]
// One wave per 8 consecutive 16-row M-tiles; 8 v_wmma_f32_16x16x32_f16 per M-tile; bias rides
// in the K=24 slot so D is stored directly (no WMMA->VALU hazard, no reg recycle).
#define TILES_PER_WAVE 8
#define M_TILES (Bn * Nn * Nn / 16)      // 102,400

__global__ void edge_wmma_kernel(const float* __restrict__ dense_e,
                                 const _Float16* __restrict__ bpack,
                                 float* __restrict__ edge_out) {
    const int lane = threadIdx.x & 31;
    const int wid  = blockIdx.x * (blockDim.x >> 5) + (threadIdx.x >> 5);
    const int m = lane & 15;      // A row within tile / D column n
    const int g = lane >> 4;      // lane group

    // Hoist B tiles (32x128 f16 incl. bias row, pre-laid-out) across the tile loop.
    v16h btile[8];
#pragma unroll
    for (int nt = 0; nt < 8; ++nt)
        btile[nt] = *(const v16h*)(bpack + ((size_t)(nt * 32 + lane)) * 16);

    for (int t = 0; t < TILES_PER_WAVE; ++t) {
        int tile = wid * TILES_PER_WAVE + t;
        if (tile >= M_TILES) break;
        int r  = tile * 16 + m;            // flattened (b,i,j) row for this lane
        int j  = r % Nn;
        int tt = r / Nn;
        int i  = tt % Nn;
        int b  = tt / Nn;
        const float4* pij = (const float4*)(dense_e + (((size_t)b * Nn + i) * Nn + j) * DINn);
        const float4* pji = (const float4*)(dense_e + (((size_t)b * Nn + j) * Nn + i) * DINn);
        float4 ij0 = pij[0], ij1 = pij[1];
        float4 ji0 = pji[0], ji1 = pji[1];
        float fij[8] = {ij0.x, ij0.y, ij0.z, ij0.w, ij1.x, ij1.y, ij1.z, ij1.w};
        float fji[8] = {ji0.x, ji0.y, ji0.z, ji0.w, ji1.x, ji1.y, ji1.z, ji1.w};

        // A 16x32 f16 layout:
        //   g=0: slots[0:8] = K0..7  (e_ij), slots[8:16] = K16..23 (e_ij + e_ji)
        //   g=1: slots[0:8] = K8..15 (e_ji), slot 8 = K24 = 1.0 (bias), rest 0
        v16h a;
#pragma unroll
        for (int k = 0; k < 8; ++k)
            a[k] = (_Float16)((g == 0) ? fij[k] : fji[k]);
#pragma unroll
        for (int k = 0; k < 8; ++k) {
            float hi = (g == 0) ? (fij[k] + fji[k]) : ((k == 0) ? 1.f : 0.f);
            a[8 + k] = (_Float16)hi;
        }

        float* outp = edge_out + (size_t)tile * 16 * DEn;
#pragma unroll
        for (int nt = 0; nt < 8; ++nt) {
            v8f c = {};    // inline-zero accumulator -> v_wmma ..., 0
            c = __builtin_amdgcn_wmma_f32_16x16x32_f16(
                    false, a, false, btile[nt], (short)0, c, false, false);
            // D layout: lane column n = lane%16, VGPR v -> row M = v + 8*g
#pragma unroll
            for (int v = 0; v < 8; ++v)
                __builtin_nontemporal_store(
                    c[v], &outp[(size_t)(v + 8 * g) * DEn + nt * 16 + m]);
        }
    }
}

extern "C" void kernel_launch(void* const* d_in, const int* in_sizes, int n_in,
                              void* d_out, int out_size, void* d_ws, size_t ws_size,
                              hipStream_t stream) {
    const float* x          = (const float*)d_in[0];
    const float* edge_attr  = (const float*)d_in[1];
    const float* w_bias_w   = (const float*)d_in[2];
    const float* w_bias_b   = (const float*)d_in[3];
    const float* w_weight_w = (const float*)d_in[4];
    const float* w_weight_b = (const float*)d_in[5];
    const float* w_node_w   = (const float*)d_in[6];
    const float* w_node_b   = (const float*)d_in[7];
    const float* w_edge_w   = (const float*)d_in[8];
    const float* w_edge_b   = (const float*)d_in[9];
    const int*   edge_index = (const int*)d_in[10];
    const int*   node_batch = (const int*)d_in[11];

    char* ws = (char*)d_ws;
    float*    dense_e = (float*)(ws + DENSE_E_OFF);
    float*    dense_x = (float*)(ws + DENSE_X_OFF);
    int*      starts  = (int*)(ws + STARTS_OFF);
    int*      counts  = (int*)(ws + COUNTS_OFF);
    _Float16* bpack   = (_Float16*)(ws + BPACK_OFF);
    float*    b_comb  = (float*)(ws + BCOMB_OFF);
    float*    wn_comb = (float*)(ws + WNC_OFF);
    float*    bn_comb = (float*)(ws + BNC_OFF);

    float* out = (float*)d_out;

    // 1) zero dense scratch (dense_e + dense_x contiguous)
    int n4 = (int)((DENSE_X_OFF + (size_t)Bn * Nn * DINn * 4) / 16);
    zero_f4_kernel<<<2048, 256, 0, stream>>>((float4*)dense_e, n4);
    // 2) segment starts / counts
    starts_kernel<<<1, 256, 0, stream>>>(node_batch, starts, counts);
    // 3) scatter node features
    scatter_x_kernel<<<TOTALn / 256, 256, 0, stream>>>(x, node_batch, starts, dense_x);
    // 4) scatter-add edge features into dense adjacency
    scatter_edges_kernel<<<En / 256, 256, 0, stream>>>(edge_attr, edge_index, node_batch, starts, dense_e);
    // 5) fold linear layers + pack WMMA B matrix (weights + bias row)
    prep_weights_kernel<<<1, 128, 0, stream>>>(w_weight_w, w_weight_b, w_edge_w, w_edge_b,
                                               w_bias_w, w_bias_b, w_node_w, w_node_b,
                                               bpack, b_comb, wn_comb, bn_comb);
    // 6) mask (symmetrized) and node_mask
    mask_kernel<<<(int)(MASK_SZ / 256), 256, 0, stream>>>(dense_e, out + MASK_OFF);
    node_mask_kernel<<<(Bn * Nn) / 256, 256, 0, stream>>>(counts, out + NM_OFF);
    // 7) fused node path
    node_kernel<<<(int)(NODE_SZ / 256), 256, 0, stream>>>(dense_x, wn_comb, bn_comb, out + NODE_OFF);
    // 8) fused edge GEMM via WMMA (bandwidth-bound: ~839 MB NT stores)
    int waves = M_TILES / TILES_PER_WAVE;            // 12,800
    edge_wmma_kernel<<<waves / 8, 256, 0, stream>>>(dense_e, bpack, out + EDGE_OFF);
}
